// ElectronicEmbedding_37417755082828
// MI455X (gfx1250) — matrix operand order repaired and checked by reading, not compiled, not run on hardware
//
#include <hip/hip_runtime.h>

// ---------------------------------------------------------------------------
// ElectronicEmbedding for MI455X (gfx1250, wave32, WMMA).
// 4 GEMMs [200000,256]x[256,256] in bf16 WMMA (f32 accum), everything else
// fused so only e_z is read and e_psi written to HBM (+ tiny side arrays).
// ---------------------------------------------------------------------------

typedef __attribute__((ext_vector_type(16))) __bf16 v16bf;
typedef __attribute__((ext_vector_type(8)))  float  v8f;
typedef __attribute__((ext_vector_type(8)))  int    v8i;

#define FEAT 256
#define MT   16     // atom rows per workgroup
#define AS   264    // LDS stride (halfs) for bf16 tile (pad vs bank conflicts)
#define FS   260    // LDS stride (floats) for f32 tiles

__device__ __forceinline__ unsigned short f2bf(float f) {
  unsigned int u = __builtin_bit_cast(unsigned int, f);
  u += 0x7FFFu + ((u >> 16) & 1u);           // round-to-nearest-even
  return (unsigned short)(u >> 16);
}

__device__ __forceinline__ float swishf(float x, float a, float b) {
  // a * x * sigmoid(b*x)
  return a * x / (1.0f + __expf(-b * x));
}

// Wave-level: [16 x 256] (LDS, bf16) times W^T where W is [256 x 256] bf16
// row-major (row j == output column j). Produces two 16x16 f32 tiles for
// output columns (2*wid+t)*16 + (lane&15).
// A frag layout (16-bit 16x32): lane m=lane&15; koffA=0 (lanes 0-15) / 8
// (lanes 16-31); VGPR r<4 -> K=koffA+2r(,+1); r>=4 -> K=16+koffA+2(r-4)(,+1).
// B frag layout (16-bit 32x16): n=lane&15; same K mapping as A (koffA, VGPR v<4 -> K=koffA+2v(,+1); v>=4 -> K=16+koffA+2(v-4)(,+1)).
__device__ __forceinline__ void wave_gemm_16x256(
    const unsigned short* __restrict__ Abf,   // LDS bf16 [MT][AS]
    const unsigned short* __restrict__ Wbf,   // global bf16 [256][256]
    int lane, int wid, v8f acc[2])
{
  const int m     = lane & 15;
  const int koffA = (lane & 16) ? 8  : 0;
  const int koffB = (lane & 16) ? 16 : 0;
  const unsigned short* arow  = Abf + m * AS;
  const unsigned short* b0row = Wbf + (size_t)((wid * 2 + 0) * 16 + m) * FEAT;
  const unsigned short* b1row = Wbf + (size_t)((wid * 2 + 1) * 16 + m) * FEAT;
#pragma unroll
  for (int k0 = 0; k0 < FEAT; k0 += 32) {
    v8i ai, b0, b1;
#pragma unroll
    for (int r = 0; r < 8; ++r) {
      int k = k0 + koffA + ((r < 4) ? (2 * r) : (16 + 2 * (r - 4)));
      ai[r] = *(const int*)(arow + k);        // 2 packed bf16 (K even)
    }
#pragma unroll
    for (int v = 0; v < 8; ++v) {
      int k = k0 + koffA + ((v < 4) ? (2 * v) : (16 + 2 * (v - 4)));
      b0[v] = *(const int*)(b0row + k);
      b1[v] = *(const int*)(b1row + k);
    }
    v16bf a = __builtin_bit_cast(v16bf, ai);
    acc[0] = __builtin_amdgcn_wmma_f32_16x16x32_bf16(
        false, a, false, __builtin_bit_cast(v16bf, b0), (short)0, acc[0], false, false);
    acc[1] = __builtin_amdgcn_wmma_f32_16x16x32_bf16(
        false, a, false, __builtin_bit_cast(v16bf, b1), (short)0, acc[1], false, false);
  }
}

// ---------------------------------------------------------------------------
// Prep kernels
// ---------------------------------------------------------------------------
__global__ void cvt_bf16_kernel(const float* __restrict__ src,
                                unsigned short* __restrict__ dst, int n) {
  int i = blockIdx.x * blockDim.x + threadIdx.x;
  if (i < n) dst[i] = f2bf(src[i]);
}

__global__ void prefix_kernel(const int* __restrict__ num_atoms, int n_mol,
                              int* __restrict__ offsets) {
  if (blockIdx.x == 0 && threadIdx.x == 0) {
    int acc = 0;
    for (int i = 0; i < n_mol; ++i) { offsets[i] = acc; acc += num_atoms[i]; }
    offsets[n_mol] = acc;
  }
}

__global__ void seg_kernel(const int* __restrict__ offsets,
                           const int* __restrict__ num_atoms,
                           int n_mol, int n_atoms, int* __restrict__ seg) {
  int m = blockIdx.x * blockDim.x + threadIdx.x;
  if (m >= n_mol) return;
  int s = offsets[m], c = num_atoms[m];
  for (int i = 0; i < c; ++i) {
    int idx = s + i;
    if (idx < n_atoms) seg[idx] = m;
  }
}

// ---------------------------------------------------------------------------
// Kernel 1: q = e_z @ W_lin^T + b ; arg = <q, k_sel[seg]>/16 ; num = softplus;
//           denom[mol] += num.  q stays in LDS.
// ---------------------------------------------------------------------------
__global__ __launch_bounds__(256) void qk_kernel(
    const float* __restrict__ e_z, const unsigned short* __restrict__ Wlin,
    const float* __restrict__ b_lin, const float* __restrict__ psi,
    const int* __restrict__ seg, const float* __restrict__ k_plus,
    const float* __restrict__ k_minus, float* __restrict__ num_out,
    float* __restrict__ denom, int n_atoms)
{
  __shared__ unsigned short Abf[MT * AS];
  __shared__ float Q[MT * FS];
  __shared__ float argbuf[MT];
  const int tid = threadIdx.x;
  const int atom0 = blockIdx.x * MT;

  for (int i = tid; i < MT * FEAT; i += 256) {
    int m = i >> 8, c = i & 255;
    int a = atom0 + m; if (a >= n_atoms) a = n_atoms - 1;
    Abf[m * AS + c] = f2bf(e_z[(size_t)a * FEAT + c]);
  }
  if (tid < MT) argbuf[tid] = 0.0f;
  __syncthreads();

  const int lane = tid & 31, wid = tid >> 5;
  v8f acc[2] = {};
  wave_gemm_16x256(Abf, Wlin, lane, wid, acc);
  {
    const int mbase = (lane & 16) ? 8 : 0;
    const int c0 = (wid * 2) * 16 + (lane & 15);
    const int c1 = c0 + 16;
    const float bias0 = b_lin[c0], bias1 = b_lin[c1];
#pragma unroll
    for (int r = 0; r < 8; ++r) {
      Q[(mbase + r) * FS + c0] = acc[0][r] + bias0;
      Q[(mbase + r) * FS + c1] = acc[1][r] + bias1;
    }
  }
  __syncthreads();

  {  // per-row dot with k_sel, 16 threads x 16-wide chunks per row
    const int m = tid & 15, chunk = tid >> 4;
    const int a = atom0 + m;
    float partial = 0.0f;
    if (a < n_atoms) {
      const int mol = seg[a];
      const float* kp = (psi[mol] >= 0.0f) ? k_plus : k_minus;
#pragma unroll
      for (int j = 0; j < 16; ++j) {
        int c = chunk * 16 + j;
        partial += Q[m * FS + c] * kp[c];
      }
    }
    atomicAdd(&argbuf[m], partial);   // ds_add_f32
  }
  __syncthreads();

  if (tid < MT) {
    const int a = atom0 + tid;
    if (a < n_atoms) {
      float arg = argbuf[tid] * 0.0625f;            // 1/sqrt(256)
      float nm = (arg > 20.0f) ? arg : log1pf(__expf(arg));  // softplus
      num_out[a] = nm;
      atomicAdd(&denom[seg[a]], nm);                // global_atomic_add_f32
    }
  }
}

// ---------------------------------------------------------------------------
// Kernel 2: av = (psi*num/denom) * v_sel ; 3 fused GEMMs with swish between;
//           residual add; store e_psi.
// ---------------------------------------------------------------------------
__global__ __launch_bounds__(256) void mlp_kernel(
    const float* __restrict__ psi, const int* __restrict__ seg,
    const float* __restrict__ num_in, const float* __restrict__ denom,
    const float* __restrict__ v_plus, const float* __restrict__ v_minus,
    const unsigned short* __restrict__ Wr1, const unsigned short* __restrict__ Wr2,
    const unsigned short* __restrict__ Wout,
    const float* __restrict__ alpha, const float* __restrict__ beta,
    float* __restrict__ out, int n_atoms)
{
  __shared__ float AV[MT * FS];
  __shared__ float H[MT * FS];
  __shared__ unsigned short Abf[MT * AS];
  __shared__ float a_sh[MT];
  __shared__ int pos_sh[MT];
  const int tid = threadIdx.x;
  const int atom0 = blockIdx.x * MT;

  if (tid < MT) {
    int a = atom0 + tid;
    int aa = (a < n_atoms) ? a : n_atoms - 1;
    int mol = seg[aa];
    float p = psi[mol];
    a_sh[tid] = p * num_in[aa] / denom[mol];
    pos_sh[tid] = (p >= 0.0f) ? 1 : 0;
  }
  const float al0 = alpha[0], al1 = alpha[1], al2 = alpha[2];
  const float be0 = beta[0],  be1 = beta[1],  be2 = beta[2];
  __syncthreads();

  for (int i = tid; i < MT * FEAT; i += 256) {
    int m = i >> 8, j = i & 255;
    float vsel = pos_sh[m] ? v_plus[j] : v_minus[j];
    float av = a_sh[m] * vsel;
    AV[m * FS + j] = av;
    Abf[m * AS + j] = f2bf(swishf(av, al0, be0));
  }
  __syncthreads();

  const int lane = tid & 31, wid = tid >> 5;
  const int mbase = (lane & 16) ? 8 : 0;
  const int c0 = (wid * 2) * 16 + (lane & 15);
  const int c1 = c0 + 16;

  {  // h1 = swish(av) @ W_r1^T
    v8f acc[2] = {};
    wave_gemm_16x256(Abf, Wr1, lane, wid, acc);
#pragma unroll
    for (int r = 0; r < 8; ++r) {
      H[(mbase + r) * FS + c0] = acc[0][r];
      H[(mbase + r) * FS + c1] = acc[1][r];
    }
  }
  __syncthreads();
  for (int i = tid; i < MT * FEAT; i += 256) {
    int m = i >> 8, j = i & 255;
    Abf[m * AS + j] = f2bf(swishf(H[m * FS + j], al1, be1));
  }
  __syncthreads();
  {  // h2 = swish(h1) @ W_r2^T
    v8f acc[2] = {};
    wave_gemm_16x256(Abf, Wr2, lane, wid, acc);
#pragma unroll
    for (int r = 0; r < 8; ++r) {
      H[(mbase + r) * FS + c0] = acc[0][r];
      H[(mbase + r) * FS + c1] = acc[1][r];
    }
  }
  __syncthreads();
  for (int i = tid; i < MT * FEAT; i += 256) {
    int m = i >> 8, j = i & 255;
    float res = AV[m * FS + j] + H[m * FS + j];   // residual
    Abf[m * AS + j] = f2bf(swishf(res, al2, be2));
  }
  __syncthreads();
  {  // e_psi = swish(res) @ W_out^T -> global
    v8f acc[2] = {};
    wave_gemm_16x256(Abf, Wout, lane, wid, acc);
#pragma unroll
    for (int r = 0; r < 8; ++r) {
      int a = atom0 + mbase + r;
      if (a < n_atoms) {
        out[(size_t)a * FEAT + c0] = acc[0][r];
        out[(size_t)a * FEAT + c1] = acc[1][r];
      }
    }
  }
}

// ---------------------------------------------------------------------------
extern "C" void kernel_launch(void* const* d_in, const int* in_sizes, int n_in,
                              void* d_out, int out_size, void* d_ws, size_t ws_size,
                              hipStream_t stream) {
  (void)n_in; (void)out_size; (void)ws_size;
  const float* psi   = (const float*)d_in[0];
  const float* e_z   = (const float*)d_in[1];
  const int*   natm  = (const int*)d_in[2];
  const float* W_lin = (const float*)d_in[3];
  const float* b_lin = (const float*)d_in[4];
  const float* W_r1  = (const float*)d_in[5];
  const float* W_r2  = (const float*)d_in[6];
  const float* W_out = (const float*)d_in[7];
  const float* k_p   = (const float*)d_in[8];
  const float* k_m   = (const float*)d_in[9];
  const float* v_p   = (const float*)d_in[10];
  const float* v_m   = (const float*)d_in[11];
  const float* alpha = (const float*)d_in[12];
  const float* beta  = (const float*)d_in[13];

  const int n_mol   = in_sizes[0];
  const int n_atoms = in_sizes[1] / FEAT;

  char* w = (char*)d_ws;
  auto carve = [&](size_t b) { void* p = (void*)w; w += (b + 255) & ~(size_t)255; return p; };
  int*   offsets = (int*)carve((size_t)(n_mol + 1) * sizeof(int));
  int*   seg     = (int*)carve((size_t)n_atoms * sizeof(int));
  float* numb    = (float*)carve((size_t)n_atoms * sizeof(float));
  float* denom   = (float*)carve((size_t)n_mol * sizeof(float));
  unsigned short* Wbf = (unsigned short*)carve((size_t)4 * FEAT * FEAT * sizeof(unsigned short));
  unsigned short* Wlin_bf = Wbf + 0 * FEAT * FEAT;
  unsigned short* Wr1_bf  = Wbf + 1 * FEAT * FEAT;
  unsigned short* Wr2_bf  = Wbf + 2 * FEAT * FEAT;
  unsigned short* Wout_bf = Wbf + 3 * FEAT * FEAT;

  hipMemsetAsync(denom, 0, (size_t)n_mol * sizeof(float), stream);
  hipMemsetAsync(seg, 0, (size_t)n_atoms * sizeof(int), stream);

  const int nW = FEAT * FEAT;
  cvt_bf16_kernel<<<(nW + 255) / 256, 256, 0, stream>>>(W_lin, Wlin_bf, nW);
  cvt_bf16_kernel<<<(nW + 255) / 256, 256, 0, stream>>>(W_r1,  Wr1_bf,  nW);
  cvt_bf16_kernel<<<(nW + 255) / 256, 256, 0, stream>>>(W_r2,  Wr2_bf,  nW);
  cvt_bf16_kernel<<<(nW + 255) / 256, 256, 0, stream>>>(W_out, Wout_bf, nW);

  prefix_kernel<<<1, 1, 0, stream>>>(natm, n_mol, offsets);
  seg_kernel<<<(n_mol + 255) / 256, 256, 0, stream>>>(offsets, natm, n_mol, n_atoms, seg);

  const int nblk = (n_atoms + MT - 1) / MT;   // 12500 for the reference shapes
  qk_kernel<<<nblk, 256, 0, stream>>>(e_z, Wlin_bf, b_lin, psi, seg, k_p, k_m,
                                      numb, denom, n_atoms);
  mlp_kernel<<<nblk, 256, 0, stream>>>(psi, seg, numb, denom, v_p, v_m,
                                       Wr1_bf, Wr2_bf, Wout_bf, alpha, beta,
                                       (float*)d_out, n_atoms);
}
